// SKVLAD_56478819942787
// MI455X (gfx1250) — matrix operand order repaired and checked by reading, not compiled, not run on hardware
//
#include <hip/hip_runtime.h>
#include <stdint.h>

// ---------------------------------------------------------------------------
// Problem constants (fixed by the reference's setup_inputs)
// ---------------------------------------------------------------------------
#define BB   8
#define DD   1024
#define NN   4096
#define KC   32
#define HH   512
#define PP   128
#define BN_EPS 1e-5f
#define SK_EPS 1e-3f
#define SK_INV_EPS 1000.0f

typedef __attribute__((ext_vector_type(16))) __bf16 v16bf;
typedef __attribute__((ext_vector_type(8)))  float  v8f;
typedef __attribute__((ext_vector_type(4)))  unsigned v4u;
typedef __attribute__((ext_vector_type(8)))  int      v8i;
typedef __attribute__((ext_vector_type(4)))  int      v4i;

static __device__ __forceinline__ unsigned short f2bf(float f) {
  union { float f; unsigned u; } x; x.f = f;
  unsigned r = x.u + 0x7FFFu + ((x.u >> 16) & 1u);   // RNE
  return (unsigned short)(r >> 16);
}
static __device__ __forceinline__ float bf2f(unsigned short h) {
  union { unsigned u; float f; } x; x.u = ((unsigned)h) << 16;
  return x.f;
}

// ---------------------------------------------------------------------------
// TDM: issue a 2D tile load (rows x 32 bf16, row-contiguous) into LDS.
// D# built per CDNA5 ISA:
//   group0: [1:0]=count=1, [63:32]=lds_addr, [120:64]=global byte addr,
//           [127:126]=type=2
//   group1: data_size=1 (2B), tile_dim0=32, tile_dim1=rows,
//           tensor_dim0=huge (no x OOB), tensor_dim1=rows avail (y OOB -> 0),
//           tensor_dim0_stride=row stride (elements)
// Tracked by TENSORcnt; issued by one wave, consumed after s_wait_tensorcnt
// + workgroup barrier.
// ---------------------------------------------------------------------------
static __device__ __forceinline__ void tdm_load_tile(
    unsigned lds_addr, const void* gptr, unsigned tile_rows,
    unsigned long row_stride_elems, unsigned rows_avail)
{
  unsigned long ga = (unsigned long)gptr;
  v4u g0;
  g0.x = 1u;                                        // count=1 (valid descriptor)
  g0.y = lds_addr;                                  // LDS byte address
  g0.z = (unsigned)(ga & 0xFFFFFFFFu);              // global addr [95:64]
  g0.w = (unsigned)((ga >> 32) & 0x01FFFFFFu) | (2u << 30);  // addr hi | type=2

  const unsigned td0 = 0x40000000u;                 // tensor_dim0: effectively no OOB
  v8i g1;
  g1[0] = (int)(1u << 16);                          // wg_mask=0, data_size=1 (2B)
  g1[1] = (int)((td0 & 0xFFFFu) << 16);             // barrier_addr=0 | tensor_dim0 lo
  g1[2] = (int)(((td0 >> 16) & 0xFFFFu) | ((rows_avail & 0xFFFFu) << 16));
  g1[3] = (int)(((rows_avail >> 16) & 0xFFFFu) | (32u << 16));  // tile_dim0=32
  g1[4] = (int)(tile_rows & 0xFFFFu);               // tile_dim1 | tile_dim2=0
  g1[5] = (int)(row_stride_elems & 0xFFFFFFFFu);    // tensor_dim0_stride lo
  g1[6] = (int)((row_stride_elems >> 32) & 0xFFFFu);// stride hi | dim1_stride lo=0
  g1[7] = 0;

  v4i gz = {0, 0, 0, 0};
#if __clang_major__ >= 23
  v8i gz8 = {0, 0, 0, 0, 0, 0, 0, 0};
  __builtin_amdgcn_tensor_load_to_lds(g0, g1, gz, gz, gz8, 0);
#else
  __builtin_amdgcn_tensor_load_to_lds(g0, g1, gz, gz, 0);
#endif
}

// ---------------------------------------------------------------------------
// Generic strided batched GEMM:  C = alpha * (A x B) + addc + bias[row]
//   A: bf16, element (m,k) at A[bsA*z + m*sAm + k*sAk]   (M x Kd)
//   B: bf16, element (k,n) at B[bsB*z + k*sBk + n*sBn]   (Kd x Nn)
//   C: f32,  element (m,n) at C[bsC*z + m*ldC + n]
// Block = 256 threads = 8 waves, tile 64(M) x 32(N), K-step 32, double-
// buffered LDS. Staging mode is a template parameter:
//   AM==0: A k-contiguous  -> TDM tensor_load_to_lds (async, wave0-issued)
//   AM==1: A m-contiguous  -> b128 global load + LDS scatter
//   BM==0: B k-contiguous  -> TDM tensor_load_to_lds
//   BM==1: B n-contiguous  -> b64 coalesced global load + LDS scatter
// Each wave computes a 16x16 tile via v_wmma_f32_16x16x32_bf16.
// ---------------------------------------------------------------------------
template <int AM, int BM>
__global__ __launch_bounds__(256) void skv_gemm_bf16_wmma(
    const unsigned short* __restrict__ A, long bsA, long sAm, long sAk,
    const unsigned short* __restrict__ Bm, long bsB, long sBk, long sBn,
    const float* __restrict__ bias,
    float* __restrict__ C, long bsC, long ldC,
    int M, int Nn, int Kd, float alpha, float addc)
{
  __shared__ __align__(16) unsigned short As[2][64 * 32];   // [m][k]
  __shared__ __align__(16) unsigned short Bs[2][32 * 32];   // [col][k]

  const int tid  = threadIdx.x;
  const int wave = tid >> 5;
  const int lane = tid & 31;
  const int wm = wave & 3;       // 4 waves along M (64 rows)
  const int wn = wave >> 2;      // 2 waves along N (32 cols)
  const int m0 = blockIdx.y * 64;
  const int n0 = blockIdx.x * 32;
  const long bA = (long)blockIdx.z * bsA;
  const long bB = (long)blockIdx.z * bsB;
  const long bC = (long)blockIdx.z * bsC;

  constexpr bool useTDM = (AM == 0) || (BM == 0);
  constexpr short nops = (short)((AM == 0 ? 1 : 0) + (BM == 0 ? 1 : 0));

  const unsigned ldsA[2] = { (unsigned)(uintptr_t)&As[0][0],
                             (unsigned)(uintptr_t)&As[1][0] };
  const unsigned ldsB[2] = { (unsigned)(uintptr_t)&Bs[0][0],
                             (unsigned)(uintptr_t)&Bs[1][0] };

  auto issueTiles = [&](int k0, int p) {
    if constexpr (AM == 0) {
      const unsigned short* ga = A + bA + (long)m0 * sAm + k0;
      tdm_load_tile(ldsA[p], ga, 64u, (unsigned long)sAm, (unsigned)(M - m0));
    }
    if constexpr (BM == 0) {
      const unsigned short* gb = Bm + bB + (long)n0 * sBn + k0;
      tdm_load_tile(ldsB[p], gb, 32u, (unsigned long)sBn, (unsigned)(Nn - n0));
    }
  };

  union Frag { unsigned u[8]; v16bf v; };
  v8f acc = {};

  // Prologue: kick off the first tiles via the Tensor Data Mover.
  if (useTDM && wave == 0) issueTiles(0, 0);

  int p = 0;
  for (int k0 = 0; k0 < Kd; k0 += 32, p ^= 1) {
    const bool hasNext = (k0 + 32 < Kd);

    if constexpr (useTDM) {
      if (wave == 0) {
        if (hasNext) {
          issueTiles(k0 + 32, p ^ 1);    // prefetch next into other buffer
          __builtin_amdgcn_s_wait_tensorcnt(nops);     // current done, next in flight
        } else {
          __builtin_amdgcn_s_wait_tensorcnt(0);        // drain
        }
      }
    }

    // ---------------- manual staging paths (write current buffer) ----------
    if constexpr (AM == 1) {
      // m contiguous: b128 load of 8 rows at fixed k, scatter to LDS
      int k = tid >> 3, moct = (tid & 7) << 3;
      int gm = m0 + moct; if (gm > M - 8) gm = M - 8;
      const uint4* src = (const uint4*)(A + bA + (long)(k0 + k) * sAk + gm);
      union { uint4 q; unsigned short s[8]; } u; u.q = *src;
#pragma unroll
      for (int j = 0; j < 8; ++j) As[p][(moct + j) * 32 + k] = u.s[j];
      if (hasNext)
        __builtin_prefetch(A + bA + (long)(k0 + 32 + k) * sAk + gm, 0, 1);
    }
    if constexpr (BM == 1) {
      // col contiguous: b64 coalesced load of 4 cols, scatter to LDS
      int k = tid >> 3, c4 = (tid & 7) << 2;
      int gn = n0 + c4; if (gn > Nn - 4) gn = Nn - 4;
      const uint2* src = (const uint2*)(Bm + bB + (long)(k0 + k) * sBk + gn);
      union { uint2 q; unsigned short s[4]; } u; u.q = *src;
#pragma unroll
      for (int j = 0; j < 4; ++j) Bs[p][(c4 + j) * 32 + k] = u.s[j];
      if (hasNext)
        __builtin_prefetch(Bm + bB + (long)(k0 + 32 + k) * sBk + gn, 0, 1);
    }

    __syncthreads();   // TDM data + manual stores visible to all waves

    const unsigned* Asu = (const unsigned*)As[p];
    const unsigned* Bsu = (const unsigned*)Bs[p];

    // Build A fragment (16-bit A 16x32 layout):
    // lanes 0-15: M=lane, K {0..7,16..23}; lanes 16-31: M=lane-16, K {8..15,24..31}
    Frag fa, fb;
    {
      int arow  = wm * 16 + (lane & 15);
      int abase = arow * 16 + ((lane >> 4) << 2);     // dword index into As
      fa.u[0] = Asu[abase + 0];  fa.u[1] = Asu[abase + 1];
      fa.u[2] = Asu[abase + 2];  fa.u[3] = Asu[abase + 3];
      fa.u[4] = Asu[abase + 8];  fa.u[5] = Asu[abase + 9];
      fa.u[6] = Asu[abase + 10]; fa.u[7] = Asu[abase + 11];
    }
    // B fragment (B 32x16): lanes 0-15 hold K {0..7,16..23} of col n=lane; lanes 16-31 K {8..15,24..31}
    {
      int brow  = wn * 16 + (lane & 15);
      int bbase = brow * 16 + ((lane >> 4) << 2);
#pragma unroll
      for (int j = 0; j < 8; ++j) fb.u[j] = Bsu[bbase + (j & 3) + ((j >> 2) << 3)];
    }

    acc = __builtin_amdgcn_wmma_f32_16x16x32_bf16(false, fa.v, false, fb.v,
                                                  (short)0, acc, false, false);
    __syncthreads();   // frees buffer p for the TDM issue two steps ahead
  }

  // D layout: VGPR r, lanes 0-15 -> (M=r, N=lane); lanes 16-31 -> (M=8+r, N=lane-16)
  int gcol = n0 + wn * 16 + (lane & 15);
  if (gcol < Nn) {
#pragma unroll
    for (int r = 0; r < 8; ++r) {
      int grow = m0 + wm * 16 + ((lane >> 4) << 3) + r;
      if (grow < M) {
        float v = alpha * acc[r] + addc;
        if (bias) v += bias[grow];
        C[bC + (long)grow * ldC + gcol] = v;
      }
    }
  }
}

// ---------------------------------------------------------------------------
// Elementwise / reduction helpers
// ---------------------------------------------------------------------------
__global__ void skv_cvt_bf16(const float* __restrict__ X,
                             unsigned short* __restrict__ Y, long n) {
  long i = (long)blockIdx.x * blockDim.x + threadIdx.x;
  if (i < n) Y[i] = f2bf(X[i]);
}

// BatchNorm stats over (B, N) per channel: X is [B, C, Nn]
__global__ __launch_bounds__(256) void skv_bn3_stats(
    const float* __restrict__ X, int Bn, int C, int Nn,
    float* __restrict__ meanb, float* __restrict__ varb)
{
  int c = blockIdx.x;
  long tot = (long)Bn * Nn;
  float s = 0.f, s2 = 0.f;
  for (long i = threadIdx.x; i < tot; i += 256) {
    int b = (int)(i / Nn);
    long n = i - (long)b * Nn;
    float v = X[((long)b * C + c) * Nn + n];
    s += v; s2 += v * v;
  }
  __shared__ float rs[8], rs2[8];
#pragma unroll
  for (int o = 16; o > 0; o >>= 1) { s += __shfl_xor(s, o, 32); s2 += __shfl_xor(s2, o, 32); }
  int w = threadIdx.x >> 5, lane = threadIdx.x & 31;
  if (lane == 0) { rs[w] = s; rs2[w] = s2; }
  __syncthreads();
  if (threadIdx.x == 0) {
    float a = 0.f, b2 = 0.f;
    for (int i = 0; i < 8; ++i) { a += rs[i]; b2 += rs2[i]; }
    float m = a / (float)tot;
    meanb[c] = m;
    varb[c] = b2 / (float)tot - m * m;
  }
}

__global__ void skv_bn3_apply_relu_bf16(
    const float* __restrict__ X, const float* __restrict__ g,
    const float* __restrict__ t, const float* __restrict__ meanb,
    const float* __restrict__ varb, unsigned short* __restrict__ Y,
    long total, int C, int Nn)
{
  long i = (long)blockIdx.x * blockDim.x + threadIdx.x;
  if (i >= total) return;
  int c = (int)((i / Nn) % C);
  float y = g[c] * (X[i] - meanb[c]) * rsqrtf(varb[c] + BN_EPS) + t[c];
  Y[i] = f2bf(fmaxf(y, 0.f));
}

// Softmax over K (axis=1) of log_score [B,K,N]; also stores z = logsumexp_k
__global__ void skv_softmax_k(const float* __restrict__ ls,
                              unsigned short* __restrict__ sbf,
                              float* __restrict__ z)
{
  long i = (long)blockIdx.x * blockDim.x + threadIdx.x;
  long tot = (long)BB * NN;
  if (i >= tot) return;
  int b = (int)(i / NN);
  long n = i - (long)b * NN;
  const float* p = ls + ((long)b * KC) * NN + n;
  float mx = -3.4e38f;
#pragma unroll
  for (int k = 0; k < KC; ++k) mx = fmaxf(mx, p[(long)k * NN]);
  float s = 0.f;
#pragma unroll
  for (int k = 0; k < KC; ++k) s += expf(p[(long)k * NN] - mx);
  float zz = mx + logf(s);
  z[i] = zz;
  unsigned short* q = sbf + ((long)b * KC) * NN + n;
#pragma unroll
  for (int k = 0; k < KC; ++k) q[(long)k * NN] = f2bf(expf(p[(long)k * NN] - zz));
}

// pi[b,k] = max(sum_n score[b,k,n], 1e-4)
__global__ __launch_bounds__(256) void skv_pi_reduce(
    const unsigned short* __restrict__ sbf, float* __restrict__ pi)
{
  int bk = blockIdx.x;
  const unsigned short* p = sbf + (long)bk * NN;
  float s = 0.f;
  for (int i = threadIdx.x; i < NN; i += 256) s += bf2f(p[i]);
  __shared__ float rs[8];
#pragma unroll
  for (int o = 16; o > 0; o >>= 1) s += __shfl_xor(s, o, 32);
  int w = threadIdx.x >> 5, lane = threadIdx.x & 31;
  if (lane == 0) rs[w] = s;
  __syncthreads();
  if (threadIdx.x == 0) {
    float a = 0.f;
    for (int i = 0; i < 8; ++i) a += rs[i];
    pi[bk] = fmaxf(a, 1e-4f);
  }
}

// vlad /= pi (broadcast over d); also emit bf16 copy
__global__ void skv_vlad_scale(float* __restrict__ vlad,
                               unsigned short* __restrict__ vbf,
                               const float* __restrict__ pi)
{
  long i = (long)blockIdx.x * blockDim.x + threadIdx.x;
  long tot = (long)BB * DD * KC;
  if (i >= tot) return;
  int k = (int)(i % KC);
  int b = (int)(i / ((long)DD * KC));
  float v = vlad[i] / pi[b * KC + k];
  vlad[i] = v;
  vbf[i] = f2bf(v);
}

// Column inverse-L2-norm over d of X [B, D, Nn] -> inv[b*Nn+n]
__global__ void skv_colnorm_inv(const float* __restrict__ X,
                                float* __restrict__ inv, int D, int Nn, long cols)
{
  long i = (long)blockIdx.x * blockDim.x + threadIdx.x;
  if (i >= cols) return;
  int b = (int)(i / Nn);
  long n = i - (long)b * Nn;
  const float* p = X + (long)b * D * Nn + n;
  float ss = 0.f;
  for (int d = 0; d < D; ++d) { float v = p[(long)d * Nn]; ss += v * v; }
  inv[i] = 1.0f / fmaxf(sqrtf(ss), 1e-12f);
}

// Y[b,d,n] = bf16( X[b,d,n] * inv[b*Nn+n] )
__global__ void skv_colscale_bf16(const float* __restrict__ X,
                                  const float* __restrict__ inv,
                                  unsigned short* __restrict__ Y, int D, int Nn,
                                  long total)
{
  long i = (long)blockIdx.x * blockDim.x + threadIdx.x;
  if (i >= total) return;
  int b = (int)(i / ((long)D * Nn));
  long n = i % Nn;
  Y[i] = f2bf(X[i] * inv[(long)b * Nn + n]);
}

// gmax[b,d] = max_k vlad[b,d,k]
__global__ void skv_rowmax_k(const float* __restrict__ vlad,
                             float* __restrict__ gmax, long totalBD)
{
  long i = (long)blockIdx.x * blockDim.x + threadIdx.x;
  if (i >= totalBD) return;
  const float* p = vlad + i * (long)KC;
  float m = p[0];
#pragma unroll
  for (int k = 1; k < KC; ++k) m = fmaxf(m, p[k]);
  gmax[i] = m;
}

// ---------------------------------------------------------------------------
// Sinkhorn: one block per batch, 512 threads. u[4096] + v[32] live in LDS.
// ---------------------------------------------------------------------------
__global__ __launch_bounds__(512) void skv_sinkhorn(
    const float* __restrict__ cost, float* __restrict__ u_out,
    float* __restrict__ v_out)
{
  __shared__ float u_lds[NN];
  __shared__ float v_lds[KC];
  __shared__ float redU[16], redV[16];
  __shared__ float d_acc;

  const int tid = threadIdx.x;
  const int w = tid >> 5, lane = tid & 31;
  const int b = blockIdx.x;
  const float* Cb = cost + (long)b * NN * KC;
  const float logp = logf(1.0f / (float)NN + 1e-8f);
  const float logq = logf(1.0f / (float)KC + 1e-8f);

  for (int n = tid; n < NN; n += 512) u_lds[n] = 0.f;
  if (tid < KC) v_lds[tid] = 0.f;
  __syncthreads();

  for (int it = 0; it < 25; ++it) {
    // ---- u update: u += eps*(logp - lse_m((u+v-c)/eps)) ----
    float du = 0.f;
#pragma unroll
    for (int rep = 0; rep < NN / 512; ++rep) {
      int n = tid + rep * 512;
      float un = u_lds[n];
      float cr[KC];
#pragma unroll
      for (int m = 0; m < KC; ++m) cr[m] = v_lds[m] - Cb[(long)n * KC + m];
      float mx = -3.4e38f;
#pragma unroll
      for (int m = 0; m < KC; ++m) mx = fmaxf(mx, cr[m]);
      float s = 0.f;
#pragma unroll
      for (int m = 0; m < KC; ++m) s += expf((cr[m] - mx) * SK_INV_EPS);
      float lse = (un + mx) * SK_INV_EPS + logf(s);
      float unew = SK_EPS * (logp - lse) + un;
      du += fabsf(unew - un);
      u_lds[n] = unew;
    }
#pragma unroll
    for (int o = 16; o > 0; o >>= 1) du += __shfl_xor(du, o, 32);
    if (lane == 0) redU[w] = du;
    __syncthreads();

    // ---- v update: wave w owns clusters m = w and m = w+16 ----
    float dv = 0.f;
#pragma unroll
    for (int rep = 0; rep < 2; ++rep) {
      int m = w + rep * 16;
      float vm = v_lds[m];
      float mx = -3.4e38f, sm = 0.f;
      for (int n = lane; n < NN; n += 32) {
        float t = (u_lds[n] + vm - Cb[(long)n * KC + m]) * SK_INV_EPS;
        float nm = fmaxf(mx, t);
        sm = sm * expf(mx - nm) + expf(t - nm);
        mx = nm;
      }
#pragma unroll
      for (int o = 1; o < 32; o <<= 1) {
        float omx = __shfl_xor(mx, o, 32);
        float osm = __shfl_xor(sm, o, 32);
        float nm = fmaxf(mx, omx);
        sm = sm * expf(mx - nm) + osm * expf(omx - nm);
        mx = nm;
      }
      float lse = mx + logf(sm);
      float vnew = SK_EPS * (logq - lse) + vm;
      if (lane == 0) { v_lds[m] = vnew; dv += fabsf(vnew - vm); }
    }
    if (lane == 0) redV[w] = dv;
    __syncthreads();
    if (tid == 0) {
      float s = 0.f;
      for (int i = 0; i < 16; ++i) s += redU[i] + redV[i];
      d_acc = s;
    }
    __syncthreads();
    float d = d_acc;
    __syncthreads();
    if (d < 1e-3f) break;
  }

  for (int n = tid; n < NN; n += 512) u_out[(long)b * NN + n] = u_lds[n];
  if (tid < KC) v_out[b * KC + tid] = v_lds[tid];
}

// ---------------------------------------------------------------------------
// Loss: partial[blk] = sum over (b,n) of sum_k N*gamma[n,k]*(ls[k,n]-z[n])
// ---------------------------------------------------------------------------
__global__ __launch_bounds__(256) void skv_loss_partial(
    const float* __restrict__ cost, const float* __restrict__ u,
    const float* __restrict__ v, const float* __restrict__ ls,
    const float* __restrict__ z, float* __restrict__ part)
{
  long i = (long)blockIdx.x * 256 + threadIdx.x;
  long tot = (long)BB * NN;
  float p = 0.f;
  if (i < tot) {
    int b = (int)(i / NN);
    long n = i - (long)b * NN;
    float uu = u[i], zz = z[i];
    const float* cr = cost + i * (long)KC;
    const float* vb = v + b * KC;
    const float* lsb = ls + ((long)b * KC) * NN + n;
#pragma unroll
    for (int k = 0; k < KC; ++k) {
      float g = expf((uu + vb[k] - cr[k]) * SK_INV_EPS);
      p += ((float)NN * g) * (lsb[(long)k * NN] - zz);
    }
  }
  __shared__ float rs[8];
#pragma unroll
  for (int o = 16; o > 0; o >>= 1) p += __shfl_xor(p, o, 32);
  int w = threadIdx.x >> 5, lane = threadIdx.x & 31;
  if (lane == 0) rs[w] = p;
  __syncthreads();
  if (threadIdx.x == 0) {
    float s = 0.f;
    for (int j = 0; j < 8; ++j) s += rs[j];
    part[blockIdx.x] = s;
  }
}

__global__ __launch_bounds__(256) void skv_loss_final(
    const float* __restrict__ part, int nparts, float* __restrict__ out)
{
  float s = (threadIdx.x < nparts) ? part[threadIdx.x] : 0.f;
  __shared__ float rs[8];
#pragma unroll
  for (int o = 16; o > 0; o >>= 1) s += __shfl_xor(s, o, 32);
  int w = threadIdx.x >> 5, lane = threadIdx.x & 31;
  if (lane == 0) rs[w] = s;
  __syncthreads();
  if (threadIdx.x == 0) {
    float a = 0.f;
    for (int j = 0; j < 8; ++j) a += rs[j];
    out[0] = -a / (float)((long)BB * NN);
  }
}

// ---------------------------------------------------------------------------
// Tiny f32 MLP pieces (predictor head; negligible FLOPs)
// ---------------------------------------------------------------------------
__global__ void skv_mlp_gemm(const float* __restrict__ X,
                             const float* __restrict__ W,
                             const float* __restrict__ bias,
                             float* __restrict__ Y, int In, int On)
{
  int i = blockIdx.x * 256 + threadIdx.x;
  if (i >= BB * On) return;
  int b = i / On, o = i - b * On;
  const float* x = X + (long)b * In;
  const float* wr = W + (long)o * In;
  float s = bias[o];
  for (int j = 0; j < In; ++j) s += x[j] * wr[j];
  Y[i] = s;
}

__global__ void skv_bn2_relu(float* __restrict__ X, const float* __restrict__ g,
                             const float* __restrict__ t, int C)
{
  int o = blockIdx.x * 256 + threadIdx.x;
  if (o >= C) return;
  float m = 0.f;
#pragma unroll
  for (int b = 0; b < BB; ++b) m += X[(long)b * C + o];
  m /= (float)BB;
  float v = 0.f;
#pragma unroll
  for (int b = 0; b < BB; ++b) { float d = X[(long)b * C + o] - m; v += d * d; }
  v /= (float)BB;
  float sc = g[o] * rsqrtf(v + BN_EPS);
#pragma unroll
  for (int b = 0; b < BB; ++b)
    X[(long)b * C + o] = fmaxf(sc * (X[(long)b * C + o] - m) + t[o], 0.f);
}

// ---------------------------------------------------------------------------
// Host launcher
// ---------------------------------------------------------------------------
extern "C" void kernel_launch(void* const* d_in, const int* in_sizes, int n_in,
                              void* d_out, int out_size, void* d_ws, size_t ws_size,
                              hipStream_t stream) {
  (void)in_sizes; (void)n_in; (void)out_size; (void)ws_size;

  const float* feat = (const float*)d_in[0];
  const float *cw1=(const float*)d_in[1], *cb1=(const float*)d_in[2],
              *cg1=(const float*)d_in[3], *ct1=(const float*)d_in[4],
              *cw2=(const float*)d_in[5], *cb2=(const float*)d_in[6],
              *cg2=(const float*)d_in[7], *ct2=(const float*)d_in[8],
              *cw3=(const float*)d_in[9], *cb3=(const float*)d_in[10];
  const float *pw1=(const float*)d_in[11], *pb1=(const float*)d_in[12],
              *pg1=(const float*)d_in[13], *pt1=(const float*)d_in[14],
              *pw2=(const float*)d_in[15], *pb2=(const float*)d_in[16],
              *pg2=(const float*)d_in[17], *pt2=(const float*)d_in[18],
              *pw3=(const float*)d_in[19], *pb3=(const float*)d_in[20];
  const float *mw1=(const float*)d_in[21], *mb1=(const float*)d_in[22],
              *mg1=(const float*)d_in[23], *mt1=(const float*)d_in[24],
              *mw2=(const float*)d_in[25], *mb2=(const float*)d_in[26],
              *mg2=(const float*)d_in[27], *mt2=(const float*)d_in[28],
              *mw3=(const float*)d_in[29], *mb3=(const float*)d_in[30];
  float* out = (float*)d_out;

  // ---- workspace bump allocator (256B aligned) ----
  size_t off = 0;
  auto alloc = [&](size_t bytes) -> void* {
    off = (off + 255) & ~(size_t)255;
    void* p = (char*)d_ws + off;
    off += bytes;
    return p;
  };
  const long fN = (long)BB * DD * NN;       // 33.5M
  const long aN = (long)BB * HH * NN;       // 16.8M
  const long lN = (long)BB * KC * NN;       // 1.05M
  const long vN = (long)BB * DD * KC;       // 262k
  const long pvN = (long)BB * HH * KC;      // 131k

  unsigned short* Fbf   = (unsigned short*)alloc(fN * 2);   // features bf16; later nf bf16
  float*          act   = (float*)alloc(aN * 4);            // conv1/conv2 f32 out
  unsigned short* actbf = (unsigned short*)alloc(aN * 2);   // bn+relu bf16
  float*          lsc   = (float*)alloc(lN * 4);            // log_score
  float*          zbuf  = (float*)alloc((long)BB * NN * 4);
  unsigned short* scbf  = (unsigned short*)alloc(lN * 2);   // softmax score bf16
  float*          pib   = (float*)alloc(BB * KC * 4);
  float*          vlad  = (float*)alloc(vN * 4);
  unsigned short* vldbf = (unsigned short*)alloc(vN * 2);
  unsigned short* nmubf = (unsigned short*)alloc(vN * 2);
  float*          invn  = (float*)alloc((long)BB * NN * 4);
  float*          invk  = (float*)alloc(BB * KC * 4);
  float*          cost  = (float*)alloc((long)BB * NN * KC * 4);
  float*          ub    = (float*)alloc((long)BB * NN * 4);
  float*          vb    = (float*)alloc(BB * KC * 4);
  float*          gmax  = (float*)alloc((long)BB * DD * 4);
  float*          m1    = (float*)alloc((long)BB * HH * 4);
  float*          m2    = (float*)alloc((long)BB * HH * 4);
  float*          pv1   = (float*)alloc(pvN * 4);
  unsigned short* pv1b  = (unsigned short*)alloc(pvN * 2);
  float*          pv2   = (float*)alloc(pvN * 4);
  unsigned short* pv2b  = (unsigned short*)alloc(pvN * 2);
  float*          meanb = (float*)alloc(HH * 4);
  float*          varb  = (float*)alloc(HH * 4);
  float*          lossp = (float*)alloc(256 * 4);
  unsigned short* cw1b  = (unsigned short*)alloc((long)HH * DD * 2);
  unsigned short* cw2b  = (unsigned short*)alloc((long)HH * HH * 2);
  unsigned short* cw3b  = (unsigned short*)alloc((long)KC * HH * 2);
  unsigned short* pw1b  = (unsigned short*)alloc((long)HH * DD * 2);
  unsigned short* pw2b  = (unsigned short*)alloc((long)HH * HH * 2);
  unsigned short* pw3b  = (unsigned short*)alloc((long)PP * HH * 2);

  auto ew = [](long n) { return dim3((unsigned)((n + 255) / 256)); };
  // Dispatch on operand layout: AM/BM template modes (see kernel comment).
  auto gemm = [&](const unsigned short* A, long bsA, long sAm, long sAk,
                  const unsigned short* Bm, long bsB, long sBk, long sBn,
                  const float* bias, float* C, long bsC, long ldC,
                  int M, int Nc, int Kd, float alpha, float addc, int batches) {
    dim3 grid((Nc + 31) / 32, (M + 63) / 64, batches);
    if (sAk == 1 && sBk == 1)
      skv_gemm_bf16_wmma<0, 0><<<grid, 256, 0, stream>>>(
          A, bsA, sAm, sAk, Bm, bsB, sBk, sBn, bias, C, bsC, ldC, M, Nc, Kd, alpha, addc);
    else if (sAk == 1)
      skv_gemm_bf16_wmma<0, 1><<<grid, 256, 0, stream>>>(
          A, bsA, sAm, sAk, Bm, bsB, sBk, sBn, bias, C, bsC, ldC, M, Nc, Kd, alpha, addc);
    else
      skv_gemm_bf16_wmma<1, 1><<<grid, 256, 0, stream>>>(
          A, bsA, sAm, sAk, Bm, bsB, sBk, sBn, bias, C, bsC, ldC, M, Nc, Kd, alpha, addc);
  };

  // 1) bf16 conversions (features + weights used by WMMA GEMMs)
  skv_cvt_bf16<<<ew(fN), 256, 0, stream>>>(feat, Fbf, fN);
  skv_cvt_bf16<<<ew((long)HH * DD), 256, 0, stream>>>(cw1, cw1b, (long)HH * DD);
  skv_cvt_bf16<<<ew((long)HH * HH), 256, 0, stream>>>(cw2, cw2b, (long)HH * HH);
  skv_cvt_bf16<<<ew((long)KC * HH), 256, 0, stream>>>(cw3, cw3b, (long)KC * HH);
  skv_cvt_bf16<<<ew((long)HH * DD), 256, 0, stream>>>(pw1, pw1b, (long)HH * DD);
  skv_cvt_bf16<<<ew((long)HH * HH), 256, 0, stream>>>(pw2, pw2b, (long)HH * HH);
  skv_cvt_bf16<<<ew((long)PP * HH), 256, 0, stream>>>(pw3, pw3b, (long)PP * HH);

  // 2) score conv head: D->H->H->Kc
  gemm(cw1b, 0, DD, 1, Fbf, (long)DD * NN, NN, 1, cb1,
       act, (long)HH * NN, NN, HH, NN, DD, 1.f, 0.f, BB);
  skv_bn3_stats<<<HH, 256, 0, stream>>>(act, BB, HH, NN, meanb, varb);
  skv_bn3_apply_relu_bf16<<<ew(aN), 256, 0, stream>>>(act, cg1, ct1, meanb, varb,
                                                      actbf, aN, HH, NN);
  gemm(cw2b, 0, HH, 1, actbf, (long)HH * NN, NN, 1, cb2,
       act, (long)HH * NN, NN, HH, NN, HH, 1.f, 0.f, BB);
  skv_bn3_stats<<<HH, 256, 0, stream>>>(act, BB, HH, NN, meanb, varb);
  skv_bn3_apply_relu_bf16<<<ew(aN), 256, 0, stream>>>(act, cg2, ct2, meanb, varb,
                                                      actbf, aN, HH, NN);
  gemm(cw3b, 0, HH, 1, actbf, (long)HH * NN, NN, 1, cb3,
       lsc, (long)KC * NN, NN, KC, NN, HH, 1.f, 0.f, BB);

  // 3) softmax over K, pi, vlad
  skv_softmax_k<<<ew((long)BB * NN), 256, 0, stream>>>(lsc, scbf, zbuf);
  skv_pi_reduce<<<BB * KC, 256, 0, stream>>>(scbf, pib);
  // vlad[d,k] = sum_n feat[d,n] * score[k,n]  (A = feat, B = score^T via strides)
  gemm(Fbf, (long)DD * NN, NN, 1, scbf, (long)KC * NN, 1, NN, nullptr,
       vlad, (long)DD * KC, KC, DD, KC, NN, 1.f, 0.f, BB);
  skv_vlad_scale<<<ew(vN), 256, 0, stream>>>(vlad, vldbf, pib);

  // 4) normalize: nf (overwrites Fbf after vlad GEMM), nmu
  skv_colnorm_inv<<<ew((long)BB * NN), 256, 0, stream>>>(feat, invn, DD, NN,
                                                         (long)BB * NN);
  skv_colscale_bf16<<<ew(fN), 256, 0, stream>>>(feat, invn, Fbf, DD, NN, fN);
  skv_colnorm_inv<<<ew((long)BB * KC), 256, 0, stream>>>(vlad, invk, DD, KC,
                                                         (long)BB * KC);
  skv_colscale_bf16<<<ew(vN), 256, 0, stream>>>(vlad, invk, nmubf, DD, KC, vN);

  // 5) cost[n,m] = 2 - 2 * sum_d nf[d,n]*nmu[d,m]  (A = nf^T via strides)
  gemm(Fbf, (long)DD * NN, 1, NN, nmubf, (long)DD * KC, KC, 1,
       nullptr, cost, (long)NN * KC, KC, NN, KC, DD, -2.f, 2.f, BB);

  // 6) Sinkhorn + loss
  skv_sinkhorn<<<BB, 512, 0, stream>>>(cost, ub, vb);
  {
    int nblk = (int)(((long)BB * NN + 255) / 256);   // 128
    skv_loss_partial<<<nblk, 256, 0, stream>>>(cost, ub, vb, lsc, zbuf, lossp);
    skv_loss_final<<<1, 256, 0, stream>>>(lossp, nblk, out);
  }

  // 7) predictor MLP on max_k(vlad): D->H->H->P  -> c_vlad at out[1 + B*P*K]
  skv_rowmax_k<<<ew((long)BB * DD), 256, 0, stream>>>(vlad, gmax, (long)BB * DD);
  skv_mlp_gemm<<<ew((long)BB * HH), 256, 0, stream>>>(gmax, mw1, mb1, m1, DD, HH);
  skv_bn2_relu<<<ew(HH), 256, 0, stream>>>(m1, mg1, mt1, HH);
  skv_mlp_gemm<<<ew((long)BB * HH), 256, 0, stream>>>(m1, mw2, mb2, m2, HH, HH);
  skv_bn2_relu<<<ew(HH), 256, 0, stream>>>(m2, mg2, mt2, HH);
  skv_mlp_gemm<<<ew((long)BB * PP), 256, 0, stream>>>(
      m2, mw3, mb3, out + 1 + (long)BB * PP * KC, HH, PP);

  // 8) projector conv on vlad: D->H->H->P -> vlad_out at out[1]
  gemm(pw1b, 0, DD, 1, vldbf, (long)DD * KC, KC, 1, pb1,
       pv1, (long)HH * KC, KC, HH, KC, DD, 1.f, 0.f, BB);
  skv_bn3_stats<<<HH, 256, 0, stream>>>(pv1, BB, HH, KC, meanb, varb);
  skv_bn3_apply_relu_bf16<<<ew(pvN), 256, 0, stream>>>(pv1, pg1, pt1, meanb, varb,
                                                       pv1b, pvN, HH, KC);
  gemm(pw2b, 0, HH, 1, pv1b, (long)HH * KC, KC, 1, pb2,
       pv2, (long)HH * KC, KC, HH, KC, HH, 1.f, 0.f, BB);
  skv_bn3_stats<<<HH, 256, 0, stream>>>(pv2, BB, HH, KC, meanb, varb);
  skv_bn3_apply_relu_bf16<<<ew(pvN), 256, 0, stream>>>(pv2, pg2, pt2, meanb, varb,
                                                       pv2b, pvN, HH, KC);
  gemm(pw3b, 0, HH, 1, pv2b, (long)HH * KC, KC, 1, pb3,
       out + 1, (long)PP * KC, KC, PP, KC, HH, 1.f, 0.f, BB);
}
